// PrimusTurboLocalMoELayer_19052474925062
// MI455X (gfx1250) — compile-verified, matrix operand order checked by reading, not run on hardware
//
#include <hip/hip_runtime.h>
#include <hip/hip_bf16.h>
#include <math.h>

#define T_TOKENS 8192
#define HIDDEN   2048
#define EXPERTS  8
#define INTER    1024
#define TILE_M   32

typedef __attribute__((ext_vector_type(16))) __bf16 v16bf;
typedef __attribute__((ext_vector_type(8)))  float  v8f;

union BF16x16 {
  v16bf v;
  unsigned short u[16];
  unsigned int   w[8];
};

__device__ __forceinline__ unsigned short f32_to_bf16(float f) {
  unsigned int u = __float_as_uint(f);
  u += 0x7FFFu + ((u >> 16) & 1u);   // round-to-nearest-even
  return (unsigned short)(u >> 16);
}

__device__ __forceinline__ float gelu_tanh(float v) {
  return 0.5f * v * (1.0f + tanhf(0.7978845608028654f * (v + 0.044715f * v * v * v)));
}

// ---------------------------------------------------------------------------
// Kernel 0: zero d_out accumulation region + workspace header
// header layout (floats): P_sum[8], f_sum[8], counts[8] (int bits)
// ---------------------------------------------------------------------------
__global__ void k_zero(float* __restrict__ out, float* __restrict__ ws_hdr) {
  size_t n = (size_t)T_TOKENS * HIDDEN;
  size_t i = (size_t)blockIdx.x * blockDim.x + threadIdx.x;
  size_t stride = (size_t)gridDim.x * blockDim.x;
  for (; i < n; i += stride) out[i] = 0.0f;
  if (blockIdx.x == 0 && threadIdx.x < 24) ws_hdr[threadIdx.x] = 0.0f;
}

// ---------------------------------------------------------------------------
// Kernel: convert x (f32 row-major) -> bf16 row-major, packed stores
// ---------------------------------------------------------------------------
__global__ void k_cvt_x(const float* __restrict__ x, unsigned int* __restrict__ xb2) {
  size_t n2 = (size_t)T_TOKENS * HIDDEN / 2;
  size_t i = (size_t)blockIdx.x * blockDim.x + threadIdx.x;
  size_t stride = (size_t)gridDim.x * blockDim.x;
  const float2* xp = (const float2*)x;
  for (; i < n2; i += stride) {
    float2 f = xp[i];
    unsigned int lo = f32_to_bf16(f.x);
    unsigned int hi = f32_to_bf16(f.y);
    xb2[i] = lo | (hi << 16);
  }
}

// ---------------------------------------------------------------------------
// Kernel: convert expert weights (f32, per-expert K x N row-major) into bf16
// WMMA B-fragment order: fragment(e, kchunk, ntile) = 32 lanes x 16 bf16 = 1KB
// contiguous. One thread per (fragment, lane); writes 32B.
// Lane/element -> (K,N) mapping per ISA 7.12.2 (bf16 32x16 B matrix):
//   N = ntile*16 + (lane & 15)
//   K = kchunk*32 + (j&7) + (j>=8 ? 16 : 0) + (lane>=16 ? 8 : 0)
// ---------------------------------------------------------------------------
__global__ void k_cvt_w(const float* __restrict__ src, unsigned short* __restrict__ dst,
                        int Kdim, int Ndim) {
  int ntiles = Ndim >> 4;
  int kcs = Kdim >> 5;
  size_t gid = (size_t)blockIdx.x * blockDim.x + threadIdx.x;
  size_t total = (size_t)EXPERTS * kcs * ntiles * 32;
  if (gid >= total) return;
  int l = (int)(gid & 31);
  size_t f = gid >> 5;
  int nt = (int)(f % ntiles);
  size_t f2 = f / ntiles;
  int kc = (int)(f2 % kcs);
  int e  = (int)(f2 / kcs);

  const float* se = src + (size_t)e * Kdim * Ndim;
  int N = nt * 16 + (l & 15);
  int kbase = kc * 32 + ((l >> 4) << 3);
  BF16x16 b;
#pragma unroll
  for (int j = 0; j < 8; ++j)
    b.u[j] = f32_to_bf16(se[(size_t)(kbase + j) * Ndim + N]);
#pragma unroll
  for (int j = 0; j < 8; ++j)
    b.u[8 + j] = f32_to_bf16(se[(size_t)(kbase + 16 + j) * Ndim + N]);

  uint4* dq = (uint4*)(dst + f * 512 + (size_t)l * 16);
  dq[0] = make_uint4(b.w[0], b.w[1], b.w[2], b.w[3]);
  dq[1] = make_uint4(b.w[4], b.w[5], b.w[6], b.w[7]);
}

// ---------------------------------------------------------------------------
// Kernel 1: router. One wave32 per token.
// ---------------------------------------------------------------------------
__global__ void __launch_bounds__(256) k_router(
    const float* __restrict__ x, const float* __restrict__ gw,
    float* __restrict__ P_sum, float* __restrict__ f_sum,
    int* __restrict__ counts, int* __restrict__ lists, float* __restrict__ wts) {
  int wave = threadIdx.x >> 5;
  int lane = threadIdx.x & 31;
  int t = blockIdx.x * (blockDim.x >> 5) + wave;
  if (t >= T_TOKENS) return;

  float xr[HIDDEN / 32];
  const float* xp = x + (size_t)t * HIDDEN;
#pragma unroll
  for (int i = 0; i < HIDDEN / 32; ++i) xr[i] = xp[lane + i * 32];

  float logit[EXPERTS];
#pragma unroll
  for (int e = 0; e < EXPERTS; ++e) {
    const float* gp = gw + (size_t)e * HIDDEN;
    float acc = 0.0f;
#pragma unroll
    for (int i = 0; i < HIDDEN / 32; ++i) acc += xr[i] * gp[lane + i * 32];
#pragma unroll
    for (int m = 16; m >= 1; m >>= 1) acc += __shfl_xor(acc, m, 32);
    logit[e] = acc;
  }

  if (lane == 0) {
    float mx = logit[0];
#pragma unroll
    for (int e = 1; e < EXPERTS; ++e) mx = fmaxf(mx, logit[e]);
    float p[EXPERTS], s = 0.0f;
#pragma unroll
    for (int e = 0; e < EXPERTS; ++e) { p[e] = __expf(logit[e] - mx); s += p[e]; }
    float inv = 1.0f / s;
    int i0 = 0;
#pragma unroll
    for (int e = 1; e < EXPERTS; ++e) if (p[e] > p[i0]) i0 = e;
    int i1 = (i0 == 0) ? 1 : 0;
#pragma unroll
    for (int e = 0; e < EXPERTS; ++e) if (e != i0 && p[e] > p[i1]) i1 = e;
    float p0 = p[i0] * inv, p1 = p[i1] * inv;
    float wnorm = 1.0f / (p0 + p1);
#pragma unroll
    for (int e = 0; e < EXPERTS; ++e) atomicAdd(&P_sum[e], p[e] * inv);
    atomicAdd(&f_sum[i0], 1.0f);
    atomicAdd(&f_sum[i1], 1.0f);
    int pos0 = atomicAdd(&counts[i0], 1);
    lists[i0 * T_TOKENS + pos0] = t;
    wts[i0 * T_TOKENS + pos0] = p0 * wnorm;
    int pos1 = atomicAdd(&counts[i1], 1);
    lists[i1 * T_TOKENS + pos1] = t;
    wts[i1 * T_TOKENS + pos1] = p1 * wnorm;
  }
}

// ---------------------------------------------------------------------------
// Kernel 2: aux loss scalar -> d_out[T*H]
// ---------------------------------------------------------------------------
__global__ void k_aux(const float* __restrict__ P_sum, const float* __restrict__ f_sum,
                      float* __restrict__ out_aux) {
  if (threadIdx.x == 0 && blockIdx.x == 0) {
    float s = 0.0f;
#pragma unroll
    for (int e = 0; e < EXPERTS; ++e)
      s += (f_sum[e] / (float)T_TOKENS) * (P_sum[e] / (float)T_TOKENS);
    *out_aux = 0.01f * (float)EXPERTS * s;
  }
}

// ---------------------------------------------------------------------------
// Kernel 3: grouped expert MLP, bf16 WMMA, pre-swizzled bf16 operands.
// grid = (T_TOKENS/TILE_M, EXPERTS), block = 256 (8 waves).
// Each block: 32 tokens of expert e (2 M-tiles of 16).
// ---------------------------------------------------------------------------
__global__ void __launch_bounds__(256) k_expert(
    const unsigned short* __restrict__ xb,    // [T][H] bf16
    const unsigned short* __restrict__ w1f,   // frag-order bf16, K=HIDDEN,N=INTER
    const unsigned short* __restrict__ w2f,   // frag-order bf16, K=INTER,N=HIDDEN
    const int* __restrict__ counts, const int* __restrict__ lists,
    const float* __restrict__ wts, float* __restrict__ out) {
  int e = blockIdx.y;
  int cnt = counts[e];
  int row0 = blockIdx.x * TILE_M;
  if (row0 >= cnt) return;

  __shared__ int   sTok[TILE_M];
  __shared__ float sWt[TILE_M];
  __shared__ unsigned short sH[TILE_M][INTER + 8];   // bf16 H tile (~66KB)

  if (threadIdx.x < TILE_M) {
    int r = row0 + threadIdx.x;
    if (r < cnt) {
      sTok[threadIdx.x] = lists[e * T_TOKENS + r];
      sWt[threadIdx.x]  = wts[e * T_TOKENS + r];
    } else {
      sTok[threadIdx.x] = 0;      // valid row, zero weight
      sWt[threadIdx.x]  = 0.0f;
    }
  }
  __syncthreads();

  const int wid  = threadIdx.x >> 5;
  const int lane = threadIdx.x & 31;
  const int hl   = lane >> 4;
  const int lm   = lane & 15;

  // ------------- GEMM1: H[32 x INTER] = X @ W1e, wave owns 128 INTER cols ----
  const unsigned short* xr0 = xb + (size_t)sTok[lm] * HIDDEN;        // M-tile 0
  const unsigned short* xr1 = xb + (size_t)sTok[16 + lm] * HIDDEN;   // M-tile 1
  // fragment stream base for this wave: frag index (e*64 + kc)*64 + wid*8 + nt
  const size_t w1base = ((size_t)e * (HIDDEN / 32)) * (INTER / 16);

  v8f acc1[2][8];
#pragma unroll
  for (int mt = 0; mt < 2; ++mt)
#pragma unroll
    for (int nt = 0; nt < 8; ++nt)
#pragma unroll
      for (int r = 0; r < 8; ++r) acc1[mt][nt][r] = 0.0f;

  for (int kc = 0; kc < HIDDEN / 32; ++kc) {
    int kk = kc * 32;
    // prefetch next k-chunk's fragment block for this wave
    if (kc + 1 < HIDDEN / 32) {
      const unsigned short* pf =
          w1f + ((w1base + (size_t)(kc + 1) * (INTER / 16)) + wid * 8) * 512;
      __builtin_prefetch(pf, 0, 1);
    }
    BF16x16 a[2];
    {
      const uint4* p00 = (const uint4*)(xr0 + kk + hl * 8);
      const uint4* p01 = (const uint4*)(xr0 + kk + 16 + hl * 8);
      const uint4* p10 = (const uint4*)(xr1 + kk + hl * 8);
      const uint4* p11 = (const uint4*)(xr1 + kk + 16 + hl * 8);
      uint4 q;
      q = *p00; a[0].w[0] = q.x; a[0].w[1] = q.y; a[0].w[2] = q.z; a[0].w[3] = q.w;
      q = *p01; a[0].w[4] = q.x; a[0].w[5] = q.y; a[0].w[6] = q.z; a[0].w[7] = q.w;
      q = *p10; a[1].w[0] = q.x; a[1].w[1] = q.y; a[1].w[2] = q.z; a[1].w[3] = q.w;
      q = *p11; a[1].w[4] = q.x; a[1].w[5] = q.y; a[1].w[6] = q.z; a[1].w[7] = q.w;
    }
#pragma unroll
    for (int nt = 0; nt < 8; ++nt) {
      size_t frag = (w1base + (size_t)kc * (INTER / 16)) + (wid * 8 + nt);
      const uint4* bq = (const uint4*)(w1f + frag * 512 + (size_t)lane * 16);
      BF16x16 b;
      uint4 q0 = bq[0], q1 = bq[1];
      b.w[0] = q0.x; b.w[1] = q0.y; b.w[2] = q0.z; b.w[3] = q0.w;
      b.w[4] = q1.x; b.w[5] = q1.y; b.w[6] = q1.z; b.w[7] = q1.w;
      acc1[0][nt] = __builtin_amdgcn_wmma_f32_16x16x32_bf16(
          false, a[0].v, false, b.v, (short)0, acc1[0][nt], false, false);
      acc1[1][nt] = __builtin_amdgcn_wmma_f32_16x16x32_bf16(
          false, a[1].v, false, b.v, (short)0, acc1[1][nt], false, false);
    }
  }

  // gelu + store H tile (D layout: lane<16 rows 0..7, lane>=16 rows 8..15)
#pragma unroll
  for (int mt = 0; mt < 2; ++mt) {
#pragma unroll
    for (int nt = 0; nt < 8; ++nt) {
      int col = wid * 128 + nt * 16 + lm;
#pragma unroll
      for (int r = 0; r < 8; ++r) {
        sH[mt * 16 + r + hl * 8][col] = f32_to_bf16(gelu_tanh(acc1[mt][nt][r]));
      }
    }
  }
  __syncthreads();

  // ------------- GEMM2: Y[32 x HIDDEN] = H @ W2e, wave owns 256 HIDDEN cols --
  const size_t w2base = ((size_t)e * (INTER / 32)) * (HIDDEN / 16);
  for (int pass = 0; pass < 2; ++pass) {
    v8f acc2[2][8];
#pragma unroll
    for (int mt = 0; mt < 2; ++mt)
#pragma unroll
      for (int nt = 0; nt < 8; ++nt)
#pragma unroll
        for (int r = 0; r < 8; ++r) acc2[mt][nt][r] = 0.0f;

    for (int kc = 0; kc < INTER / 32; ++kc) {
      int kk = kc * 32;
      if (kc + 1 < INTER / 32) {
        const unsigned short* pf =
            w2f + ((w2base + (size_t)(kc + 1) * (HIDDEN / 16)) +
                   wid * 16 + pass * 8) * 512;
        __builtin_prefetch(pf, 0, 1);
      }
      BF16x16 a[2];
      {
        const uint4* p00 = (const uint4*)&sH[lm][kk + hl * 8];
        const uint4* p01 = (const uint4*)&sH[lm][kk + 16 + hl * 8];
        const uint4* p10 = (const uint4*)&sH[16 + lm][kk + hl * 8];
        const uint4* p11 = (const uint4*)&sH[16 + lm][kk + 16 + hl * 8];
        uint4 q;
        q = *p00; a[0].w[0] = q.x; a[0].w[1] = q.y; a[0].w[2] = q.z; a[0].w[3] = q.w;
        q = *p01; a[0].w[4] = q.x; a[0].w[5] = q.y; a[0].w[6] = q.z; a[0].w[7] = q.w;
        q = *p10; a[1].w[0] = q.x; a[1].w[1] = q.y; a[1].w[2] = q.z; a[1].w[3] = q.w;
        q = *p11; a[1].w[4] = q.x; a[1].w[5] = q.y; a[1].w[6] = q.z; a[1].w[7] = q.w;
      }
#pragma unroll
      for (int nt = 0; nt < 8; ++nt) {
        size_t frag = (w2base + (size_t)kc * (HIDDEN / 16)) +
                      (wid * 16 + pass * 8 + nt);
        const uint4* bq = (const uint4*)(w2f + frag * 512 + (size_t)lane * 16);
        BF16x16 b;
        uint4 q0 = bq[0], q1 = bq[1];
        b.w[0] = q0.x; b.w[1] = q0.y; b.w[2] = q0.z; b.w[3] = q0.w;
        b.w[4] = q1.x; b.w[5] = q1.y; b.w[6] = q1.z; b.w[7] = q1.w;
        acc2[0][nt] = __builtin_amdgcn_wmma_f32_16x16x32_bf16(
            false, a[0].v, false, b.v, (short)0, acc2[0][nt], false, false);
        acc2[1][nt] = __builtin_amdgcn_wmma_f32_16x16x32_bf16(
            false, a[1].v, false, b.v, (short)0, acc2[1][nt], false, false);
      }
    }

    // weighted scatter: out[token, n] += w_token * y
#pragma unroll
    for (int mt = 0; mt < 2; ++mt) {
#pragma unroll
      for (int nt = 0; nt < 8; ++nt) {
        int n = wid * 256 + pass * 128 + nt * 16 + lm;
#pragma unroll
        for (int r = 0; r < 8; ++r) {
          int m = mt * 16 + r + hl * 8;
          atomicAdd(&out[(size_t)sTok[m] * HIDDEN + n], sWt[m] * acc2[mt][nt][r]);
        }
      }
    }
  }
}

// ---------------------------------------------------------------------------
// Host-side launcher
// ---------------------------------------------------------------------------
extern "C" void kernel_launch(void* const* d_in, const int* in_sizes, int n_in,
                              void* d_out, int out_size, void* d_ws, size_t ws_size,
                              hipStream_t stream) {
  (void)in_sizes; (void)n_in; (void)out_size; (void)ws_size;
  const float* x  = (const float*)d_in[0];   // [T, H]
  const float* gw = (const float*)d_in[1];   // [E, H]
  const float* w1 = (const float*)d_in[2];   // [E, H, I]
  const float* w2 = (const float*)d_in[3];   // [E, I, H]
  float* out = (float*)d_out;                // [T, H] + aux scalar

  // workspace layout (bytes)
  char* ws = (char*)d_ws;
  float* P_sum = (float*)ws;                              // 8 f32
  float* f_sum = P_sum + 8;                               // 8 f32
  int*   counts = (int*)(f_sum + 8);                      // 8 i32
  int*   lists  = (int*)(ws + 128);                       // E*T i32
  float* wtsbuf = (float*)(ws + 128 + EXPERTS * T_TOKENS * 4);
  size_t off = 128 + (size_t)EXPERTS * T_TOKENS * 8;
  off = (off + 255) & ~(size_t)255;
  unsigned short* xb  = (unsigned short*)(ws + off);      // T*H bf16
  off += (size_t)T_TOKENS * HIDDEN * 2;
  unsigned short* w1f = (unsigned short*)(ws + off);      // E*H*I bf16
  off += (size_t)EXPERTS * HIDDEN * INTER * 2;
  unsigned short* w2f = (unsigned short*)(ws + off);      // E*I*H bf16

  k_zero<<<4096, 256, 0, stream>>>(out, P_sum);
  k_cvt_x<<<4096, 256, 0, stream>>>(x, (unsigned int*)xb);
  k_cvt_w<<<4096, 256, 0, stream>>>(w1, w1f, HIDDEN, INTER);
  k_cvt_w<<<4096, 256, 0, stream>>>(w2, w2f, INTER, HIDDEN);

  k_router<<<T_TOKENS / 8, 256, 0, stream>>>(x, gw, P_sum, f_sum, counts, lists, wtsbuf);
  k_aux<<<1, 32, 0, stream>>>(P_sum, f_sum, out + (size_t)T_TOKENS * HIDDEN);

  dim3 grid(T_TOKENS / TILE_M, EXPERTS);
  k_expert<<<grid, 256, 0, stream>>>(xb, w1f, w2f, counts, lists, wtsbuf, out);
}
